// MultiHeadAttention_17806934409643
// MI455X (gfx1250) — compile-verified, matrix operand order checked by reading, not run on hardware
//
#include <hip/hip_runtime.h>

typedef _Float16 half_t;
typedef __attribute__((ext_vector_type(8)))  _Float16 v8h;
typedef __attribute__((ext_vector_type(16))) _Float16 v16h;
typedef __attribute__((ext_vector_type(8)))  float    v8f;
typedef __attribute__((ext_vector_type(4)))  unsigned int u32x4;
typedef __attribute__((ext_vector_type(4)))  int      i32x4;
typedef __attribute__((ext_vector_type(8)))  int      i32x8;

#define T_CTX    1024
#define D_MODEL  1024
#define N_HEADS  16
#define HEAD_DIM 64
#define B_BATCH  8
#define M_ROWS   (B_BATCH * T_CTX)   // 8192
#define NEG_BIG  (-1e9f)

typedef __attribute__((address_space(3))) void lds_void;
static __device__ __forceinline__ unsigned int lds_addr_of(void* p) {
    return (unsigned int)(uintptr_t)(lds_void*)p;
}

// Issue one TDM tile load: 2D tile of f16, tile_dim0=64 halfs/row (128B),
// rows = tile_rows, row stride in tensor = 1024 halfs, LDS rows padded to 72
// halfs (pad_interval=32 DWORDs -> code 4, pad_amount=4 DWORDs -> code 3).
static __device__ __forceinline__ void tdm_load_tile_f16(
    const half_t* gsrc, unsigned int lds_byte_off, int tile_rows, int tensor_rows)
{
    unsigned long long ga = (unsigned long long)(uintptr_t)gsrc;
    u32x4 g0;
    g0[0] = 1u;                                   // count=1, user descriptor
    g0[1] = lds_byte_off;                         // lds_addr
    g0[2] = (unsigned int)ga;                     // global_addr[31:0]
    g0[3] = (unsigned int)(ga >> 32) | (2u << 30);// global_addr[56:32] | type=2
    i32x8 g1;
    g1[0] = (1 << 16)      // data_size = 2 bytes
          | (1 << 20)      // pad_enable
          | (4 << 22)      // pad_interval: 32 DWORDs = 64 halfs
          | (3 << 25);     // pad_amount: 4 DWORDs = 8 halfs -> 72-half stride
    g1[1] = (D_MODEL & 0xFFFF) << 16;                     // tensor_dim0 lo16
    g1[2] = (D_MODEL >> 16) | ((tensor_rows & 0xFFFF) << 16); // dim0 hi | dim1 lo
    g1[3] = (tensor_rows >> 16) | (64 << 16);             // dim1 hi | tile_dim0=64
    g1[4] = tile_rows;                                    // tile_dim1 (tile_dim2=0)
    g1[5] = D_MODEL;                                      // tensor_dim0_stride lo32
    g1[6] = 0;
    g1[7] = 0;
    i32x4 z4 = {0, 0, 0, 0};
#if __clang_major__ >= 23
    i32x8 z8 = {0, 0, 0, 0, 0, 0, 0, 0};
    __builtin_amdgcn_tensor_load_to_lds(g0, g1, z4, z4, z8, 0);
#else
    __builtin_amdgcn_tensor_load_to_lds(g0, g1, z4, z4, 0);
#endif
}

// Build a WMMA f16 fragment (A or B side) from a row-major row pointer.
// Per ISA 16-bit 16x32 layout: halfs 0..7 -> K = g*8+0..7, halfs 8..15 ->
// K = 16+g*8+0..7 (g = lane>>4). A fragment = two contiguous v8h loads.
static __device__ __forceinline__ v16h make_frag(const half_t* row, int off) {
    v16h r;
    ((v8h*)&r)[0] = *(const v8h*)(row + off);
    ((v8h*)&r)[1] = *(const v8h*)(row + off + 16);
    return r;
}

// One-time f32 -> f16 conversion (x and the four weight matrices).
__global__ __launch_bounds__(256)
void convert_f32_f16(const float* __restrict__ in, half_t* __restrict__ out, int n)
{
    int i = (blockIdx.x * 256 + threadIdx.x) * 8;
    if (i < n) {
        float4 a = ((const float4*)(in + i))[0];
        float4 b = ((const float4*)(in + i))[1];
        v8h o = { (half_t)a.x, (half_t)a.y, (half_t)a.z, (half_t)a.w,
                  (half_t)b.x, (half_t)b.y, (half_t)b.z, (half_t)b.w };
        *(v8h*)(out + i) = o;
    }
}

// C = A @ W^T (+bias) (*scale), all-f16 inputs, f32 accumulate.
// Block: 128 threads (4 waves), tile 128M x 64N, K-stage 64, TDM-staged LDS.
// Each wave owns 32 rows (2 M-tiles); per stage: 16 WMMAs.
// OMODE 0: f16 out in [B,H,T,64]. OMODE 1: f32 out in [M,1024].
template<int OMODE>
__global__ __launch_bounds__(128)
void gemm_f16(const half_t* __restrict__ A, const half_t* __restrict__ W,
              const float* __restrict__ bias, float scale, void* __restrict__ Out)
{
    __shared__ __align__(16) half_t ldsA[128 * 72];
    __shared__ __align__(16) half_t ldsB[64 * 72];
    const int tid  = threadIdx.x;
    const int wave = tid >> 5;
    const int lane = tid & 31;
    const int lh   = lane & 15;
    const int g    = lane >> 4;
    const int n0   = blockIdx.x * 64;
    const int m0   = blockIdx.y * 128;

    v8f acc[2][4];
    for (int mt = 0; mt < 2; ++mt)
        for (int f = 0; f < 4; ++f)
            for (int j = 0; j < 8; ++j) acc[mt][f][j] = 0.0f;

    const unsigned int ldsA_off = lds_addr_of(ldsA);
    const unsigned int ldsB_off = lds_addr_of(ldsB);

    for (int k0 = 0; k0 < D_MODEL; k0 += 64) {
        if (wave == 0) {
            tdm_load_tile_f16(A + (size_t)m0 * D_MODEL + k0, ldsA_off, 128, M_ROWS);
            tdm_load_tile_f16(W + (size_t)n0 * D_MODEL + k0, ldsB_off, 64,  D_MODEL);
            __builtin_amdgcn_s_wait_tensorcnt(0);
        }
        __syncthreads();

        for (int ks = 0; ks < 64; ks += 32) {
            v16h a0 = make_frag(&ldsA[(wave * 32 + lh) * 72],      ks + g * 8);
            v16h a1 = make_frag(&ldsA[(wave * 32 + 16 + lh) * 72], ks + g * 8);
            for (int f = 0; f < 4; ++f) {
                v16h b = make_frag(&ldsB[(f * 16 + lh) * 72], ks + g * 8);
                acc[0][f] = __builtin_amdgcn_wmma_f32_16x16x32_f16(
                    false, a0, false, b, (short)0, acc[0][f], false, false);
                acc[1][f] = __builtin_amdgcn_wmma_f32_16x16x32_f16(
                    false, a1, false, b, (short)0, acc[1][f], false, false);
            }
        }
        __syncthreads();
    }

    // C/D layout: row = base + v + 8*g, col = n0 + f*16 + lh.
    for (int mt = 0; mt < 2; ++mt) {
        for (int f = 0; f < 4; ++f) {
            int ng = n0 + f * 16 + lh;
            float bv = bias ? bias[ng] : 0.0f;
            for (int v = 0; v < 8; ++v) {
                int   mg  = m0 + wave * 32 + mt * 16 + v + 8 * g;
                float val = (acc[mt][f][v] + bv) * scale;
                if (OMODE == 0) {
                    int b = mg >> 10, t = mg & 1023;
                    int h = ng >> 6,  d = ng & 63;
                    ((half_t*)Out)[(size_t)b * (T_CTX * D_MODEL) +
                                   (size_t)h * (T_CTX * HEAD_DIM) + t * HEAD_DIM + d] =
                        (half_t)val;
                } else {
                    ((float*)Out)[(size_t)mg * D_MODEL + ng] = val;
                }
            }
        }
    }
}

// Flash-style causal attention per (b,h). Q,K already scaled by 64^-0.25.
// Grid: (T/64, B*H); 128 threads; wave w owns q rows [qb+16w, qb+16w+16).
__global__ __launch_bounds__(128)
void attn(const half_t* __restrict__ Q, const half_t* __restrict__ Kv,
          const half_t* __restrict__ V, half_t* __restrict__ O)
{
    __shared__ __align__(16) half_t ldsVt[64 * 72];      // V tile transposed [d][key]
    __shared__ __align__(16) half_t ldsP[4][16 * 64];    // per-wave P tile [m][key]
    const int tid  = threadIdx.x;
    const int wave = tid >> 5;
    const int lane = tid & 31;
    const int lh   = lane & 15;
    const int g    = lane >> 4;
    const int qb   = blockIdx.x * 64;
    const int bh   = blockIdx.y;                         // b*H + h
    const size_t base = (size_t)bh * T_CTX * HEAD_DIM;
    const int q0 = qb + wave * 16;

    v16h qa[2];
    qa[0] = make_frag(Q + base + (size_t)(q0 + lh) * HEAD_DIM, 0 * 32 + g * 8);
    qa[1] = make_frag(Q + base + (size_t)(q0 + lh) * HEAD_DIM, 1 * 32 + g * 8);

    v8f accO[4];
    float rmax[8], rsum[8];
    for (int f = 0; f < 4; ++f)
        for (int j = 0; j < 8; ++j) accO[f][j] = 0.0f;
    for (int j = 0; j < 8; ++j) { rmax[j] = -3.0e38f; rsum[j] = 0.0f; }

    for (int kb = 0; kb <= qb; kb += 64) {
        for (int idx = tid; idx < 64 * 64; idx += 128) {
            int key = idx >> 6, d = idx & 63;
            ldsVt[d * 72 + key] = V[base + (size_t)(kb + key) * HEAD_DIM + d];
        }
        __syncthreads();

        // S = Q K^T : K fragments straight from global (row-major [t][d]).
        v8f s[4];
        for (int f = 0; f < 4; ++f) {
            for (int j = 0; j < 8; ++j) s[f][j] = 0.0f;
            for (int ks = 0; ks < 2; ++ks) {
                v16h kf = make_frag(Kv + base + (size_t)(kb + f * 16 + lh) * HEAD_DIM,
                                    ks * 32 + g * 8);
                s[f] = __builtin_amdgcn_wmma_f32_16x16x32_f16(
                    false, qa[ks], false, kf, (short)0, s[f], false, false);
            }
        }

        if (kb == qb) {  // additive causal mask, diagonal block only
            for (int f = 0; f < 4; ++f)
                for (int v = 0; v < 8; ++v) {
                    int row = q0 + v + 8 * g;
                    int key = kb + f * 16 + lh;
                    if (key > row) s[f][v] += NEG_BIG;
                }
        }

        float bmax[8];
        for (int v = 0; v < 8; ++v) {
            float m = fmaxf(fmaxf(s[0][v], s[1][v]), fmaxf(s[2][v], s[3][v]));
            for (int off = 1; off < 16; off <<= 1)
                m = fmaxf(m, __shfl_xor(m, off, 32));
            bmax[v] = m;
        }
        for (int v = 0; v < 8; ++v) {
            float nm   = fmaxf(rmax[v], bmax[v]);
            float corr = __expf(rmax[v] - nm);
            rmax[v] = nm;
            rsum[v] *= corr;
            for (int f = 0; f < 4; ++f) accO[f][v] *= corr;
        }
        for (int v = 0; v < 8; ++v) {
            float sum = 0.0f;
            for (int f = 0; f < 4; ++f) {
                float p = __expf(s[f][v] - rmax[v]);
                sum += p;
                ldsP[wave][(v + 8 * g) * 64 + f * 16 + lh] = (half_t)p;
            }
            for (int off = 1; off < 16; off <<= 1)
                sum += __shfl_xor(sum, off, 32);
            rsum[v] += sum;
        }

        for (int ks = 0; ks < 2; ++ks) {
            v16h pa = make_frag(&ldsP[wave][lh * 64], ks * 32 + g * 8);
            for (int f = 0; f < 4; ++f) {
                v16h vb = make_frag(&ldsVt[(f * 16 + lh) * 72], ks * 32 + g * 8);
                accO[f] = __builtin_amdgcn_wmma_f32_16x16x32_f16(
                    false, pa, false, vb, (short)0, accO[f], false, false);
            }
        }
        __syncthreads();   // protect ldsVt before next block's load
    }

    const int b = bh >> 4, h = bh & 15;
    for (int v = 0; v < 8; ++v) {
        float inv = 1.0f / rsum[v];
        int t = q0 + v + 8 * g;
        for (int f = 0; f < 4; ++f) {
            int col = h * HEAD_DIM + f * 16 + lh;
            O[(size_t)(b * T_CTX + t) * D_MODEL + col] = (half_t)(accO[f][v] * inv);
        }
    }
}

extern "C" void kernel_launch(void* const* d_in, const int* in_sizes, int n_in,
                              void* d_out, int out_size, void* d_ws, size_t ws_size,
                              hipStream_t stream) {
    (void)in_sizes; (void)n_in; (void)out_size; (void)ws_size;
    const float* x  = (const float*)d_in[0];
    // d_in[1] = additive causal mask; reproduced analytically (triu(-1e9, k=1)).
    const float* Wq = (const float*)d_in[2];
    const float* bq = (const float*)d_in[3];
    const float* Wk = (const float*)d_in[4];
    const float* Wv = (const float*)d_in[5];
    const float* bv = (const float*)d_in[6];
    const float* Wo = (const float*)d_in[7];
    const float* bo = (const float*)d_in[8];

    const size_t NX = (size_t)M_ROWS * D_MODEL;   // 8M
    const size_t NW = (size_t)D_MODEL * D_MODEL;  // 1M
    half_t* xh  = (half_t*)d_ws;          // 16 MB
    half_t* Wqh = xh  + NX;               // 2 MB each
    half_t* Wkh = Wqh + NW;
    half_t* Wvh = Wkh + NW;
    half_t* Woh = Wvh + NW;
    half_t* Qh  = Woh + NW;               // 16 MB each
    half_t* Kh  = Qh  + NX;
    half_t* Vh  = Kh  + NX;
    half_t* Oh  = Vh  + NX;               // total ws: 88 MB

    convert_f32_f16<<<(int)(NX / 8 / 256), 256, 0, stream>>>(x,  xh,  (int)NX);
    convert_f32_f16<<<(int)(NW / 8 / 256), 256, 0, stream>>>(Wq, Wqh, (int)NW);
    convert_f32_f16<<<(int)(NW / 8 / 256), 256, 0, stream>>>(Wk, Wkh, (int)NW);
    convert_f32_f16<<<(int)(NW / 8 / 256), 256, 0, stream>>>(Wv, Wvh, (int)NW);
    convert_f32_f16<<<(int)(NW / 8 / 256), 256, 0, stream>>>(Wo, Woh, (int)NW);

    const float scale = 0.35355339059327379f;     // 64^(-1/4)
    dim3 ggrid(D_MODEL / 64, M_ROWS / 128);
    gemm_f16<0><<<ggrid, 128, 0, stream>>>(xh, Wqh, bq,      scale, Qh);
    gemm_f16<0><<<ggrid, 128, 0, stream>>>(xh, Wkh, nullptr, scale, Kh);
    gemm_f16<0><<<ggrid, 128, 0, stream>>>(xh, Wvh, bv,      1.0f,  Vh);
    attn<<<dim3(T_CTX / 64, B_BATCH * N_HEADS), 128, 0, stream>>>(Qh, Kh, Vh, Oh);
    gemm_f16<1><<<ggrid, 128, 0, stream>>>(Oh, Woh, bo, 1.0f, d_out);
}
